// TL_20255065767968
// MI455X (gfx1250) — matrix operand order repaired and checked by reading, not compiled, not run on hardware
//
#include <hip/hip_runtime.h>
#include <cstdint>

// ---- types ----
typedef __attribute__((ext_vector_type(16))) _Float16 v16h;
typedef __attribute__((ext_vector_type(8)))  float    v8f;
typedef __attribute__((ext_vector_type(4)))  float    f32x4;
typedef __attribute__((ext_vector_type(4)))  _Float16 h4;
typedef __attribute__((ext_vector_type(4)))  unsigned u32x4;
typedef __attribute__((ext_vector_type(8)))  int      i32x8;
typedef __attribute__((ext_vector_type(4)))  int      i32x4;

// Problem sizes (fixed by the reference)
constexpr int KDIM = 4096;   // I
constexpr int NDIM = 4096;   // O
constexpr int MDIM = 16384;  // 8 * 2048

// GEMM tiling
constexpr int BK = 64;       // k-slice staged per TDM transfer (2 WMMA k-chunks)
constexpr int BM = 128;      // block rows   (2 waves x 64)
constexpr int BN = 256;      // block cols   (4 waves x 64)

// ================= kernel 0: per-block partial sums of |W| =================
__global__ void k_abs_partial(const float* __restrict__ W, float* __restrict__ partial) {
  const int tid = threadIdx.x;
  const int gid = blockIdx.x * blockDim.x + tid;
  const int nthreads = gridDim.x * blockDim.x;       // 1024 * 256
  const f32x4* W4 = (const f32x4*)W;
  const int n4 = (NDIM * KDIM) / 4;
  float s = 0.f;
  for (int i = gid; i < n4; i += nthreads) {
    f32x4 v = W4[i];
    s += fabsf(v.x) + fabsf(v.y) + fabsf(v.z) + fabsf(v.w);
  }
  __shared__ float red[256];
  red[tid] = s;
  __syncthreads();
  for (int off = 128; off > 0; off >>= 1) {
    if (tid < off) red[tid] += red[tid + off];
    __syncthreads();
  }
  if (tid == 0) partial[blockIdx.x] = red[0];
}

// ================= kernel 1: final reduce -> alpha = 0.5 * mean|W| =========
__global__ void k_alpha(const float* __restrict__ partial, float* __restrict__ alpha) {
  const int tid = threadIdx.x;  // 256 threads, 1024 partials
  float s = partial[tid] + partial[tid + 256] + partial[tid + 512] + partial[tid + 768];
  __shared__ float red[256];
  red[tid] = s;
  __syncthreads();
  for (int off = 128; off > 0; off >>= 1) {
    if (tid < off) red[tid] += red[tid + off];
    __syncthreads();
  }
  if (tid == 0) alpha[0] = 0.5f * red[0] / (float)((size_t)NDIM * KDIM);
}

// ================= kernel 2: ternary-quantize W -> fp16 {-1,0,1} ===========
__global__ void k_quant_w(const float* __restrict__ W, const float* __restrict__ alpha,
                          _Float16* __restrict__ Wq) {
  const float a = alpha[0];
  const int i = blockIdx.x * blockDim.x + threadIdx.x;   // 4 elems per thread
  f32x4 v = ((const f32x4*)W)[i];
  h4 o;
  o.x = (_Float16)((fabsf(v.x) > a) ? (v.x > 0.f ? 1.f : -1.f) : 0.f);
  o.y = (_Float16)((fabsf(v.y) > a) ? (v.y > 0.f ? 1.f : -1.f) : 0.f);
  o.z = (_Float16)((fabsf(v.z) > a) ? (v.z > 0.f ? 1.f : -1.f) : 0.f);
  o.w = (_Float16)((fabsf(v.w) > a) ? (v.w > 0.f ? 1.f : -1.f) : 0.f);
  ((h4*)Wq)[i] = o;
}

// ================= kernel 3: X fp32 -> fp16 ================================
__global__ void k_cvt_x(const float* __restrict__ X, _Float16* __restrict__ Xh) {
  const int i = blockIdx.x * blockDim.x + threadIdx.x;   // 4 elems per thread
  f32x4 v = ((const f32x4*)X)[i];
  h4 o;
  o.x = (_Float16)v.x;
  o.y = (_Float16)v.y;
  o.z = (_Float16)v.z;
  o.w = (_Float16)v.w;
  ((h4*)Xh)[i] = o;
}

// ================= TDM helper: async 2-D tile load, fp16 elements ==========
// Builds the D# per CDNA5 ISA sec 8.3/8.4 and issues TENSOR_LOAD_TO_LDS.
// tile: tile_d0 (contiguous, along K) x tile_d1 (rows); tensor row stride = stride0.
__device__ __forceinline__ void tdm_load_2d(uint32_t lds_byte_off, uint64_t gaddr,
                                            uint32_t tensor_d0, uint32_t tensor_d1,
                                            uint32_t tile_d0, uint32_t tile_d1,
                                            uint32_t stride0) {
  u32x4 g0;
  g0[0] = 1u;                                               // count=1, user mode
  g0[1] = lds_byte_off;                                     // lds_addr [63:32]
  g0[2] = (uint32_t)gaddr;                                  // global_addr [95:64]
  g0[3] = (uint32_t)((gaddr >> 32) & 0x01FFFFFFu)           // global_addr [120:96]
        | (2u << 30);                                       // type=2 ("image")
  i32x8 g1;
  g1[0] = (int)(1u << 16);                                  // data_size=1 (2 bytes)
  g1[1] = (int)((tensor_d0 & 0xFFFFu) << 16);               // tensor_dim0[15:0] @ bits 63:48
  g1[2] = (int)((tensor_d0 >> 16) | ((tensor_d1 & 0xFFFFu) << 16)); // d0 hi | d1 lo
  g1[3] = (int)((tensor_d1 >> 16) | (tile_d0 << 16));       // d1 hi | tile_dim0
  g1[4] = (int)(tile_d1);                                   // tile_dim1 (tile_dim2=0)
  g1[5] = (int)stride0;                                     // tensor_dim0_stride lo32
  g1[6] = 0;                                                // stride0 hi | stride1 lo (unused, 2-D)
  g1[7] = 0;
  i32x4 z4 = {0, 0, 0, 0};
#if __clang_major__ >= 23
  i32x8 z8 = {0, 0, 0, 0, 0, 0, 0, 0};
  __builtin_amdgcn_tensor_load_to_lds(g0, g1, z4, z4, z8, 0);
#else
  __builtin_amdgcn_tensor_load_to_lds(g0, g1, z4, z4, 0);
#endif
}

// ================= kernel 4: WMMA GEMM Y = Xh * Wq^T + b ===================
// Block: 256 threads = 8 waves (2 x 4). Wave tile 64x64 = 4x4 WMMA fragments.
// Block tile 128x256. K staged 64-wide through LDS by the Tensor Data Mover,
// double-buffered (96 KB); wave 0 issues DMAs, TENSORcnt + barriers sync.
// 32 v_wmma per barrier window.
__global__ void __launch_bounds__(256, 1)
k_gemm(const _Float16* __restrict__ Xh, const _Float16* __restrict__ Wq,
       const float* __restrict__ bias, float* __restrict__ Y) {
  __shared__ _Float16 sX[2][BM * BK];   // 2 x 16 KB
  __shared__ _Float16 sW[2][BN * BK];   // 2 x 32 KB

  const int lane = threadIdx.x & 31;
  const int wv   = threadIdx.x >> 5;    // 0..7
  const int wm   = wv >> 2;             // 0..1
  const int wn   = wv & 3;              // 0..3
  const int mblk = blockIdx.y * BM;
  const int nblk = blockIdx.x * BN;

  const uint64_t xbase = (uint64_t)(uintptr_t)Xh + 2ull * ((uint64_t)mblk * KDIM);
  const uint64_t wbase = (uint64_t)(uintptr_t)Wq + 2ull * ((uint64_t)nblk * KDIM);
  // LDS byte offsets: generic LDS address low 32 bits == LDS offset (ISA 10.2)
  const uint32_t sx0 = (uint32_t)(uintptr_t)&sX[0][0];
  const uint32_t sx1 = (uint32_t)(uintptr_t)&sX[1][0];
  const uint32_t sw0 = (uint32_t)(uintptr_t)&sW[0][0];
  const uint32_t sw1 = (uint32_t)(uintptr_t)&sW[1][0];

  v8f acc[4][4] = {};

  const int nsteps = KDIM / BK;   // 64
  if (wv == 0) {                  // prologue: stage k=0 into buffer 0
    tdm_load_2d(sx0, xbase, KDIM, MDIM, BK, BM, KDIM);
    tdm_load_2d(sw0, wbase, KDIM, NDIM, BK, BN, KDIM);
  }

  for (int s = 0; s < nsteps; ++s) {
    const int p = s & 1;
    __syncthreads();              // all waves done reading buffer p^1 (from step s-1)
    if (wv == 0) {
      if (s + 1 < nsteps) {       // prefetch next k-slice into the other buffer
        const uint64_t koff = 2ull * (uint64_t)(s + 1) * BK;
        tdm_load_2d(p ? sx0 : sx1, xbase + koff, KDIM, MDIM, BK, BM, KDIM);
        tdm_load_2d(p ? sw0 : sw1, wbase + koff, KDIM, NDIM, BK, BN, KDIM);
        __builtin_amdgcn_s_wait_tensorcnt(2);  // in-order: oldest 2 (buffer p) done
      } else {
        __builtin_amdgcn_s_wait_tensorcnt(0);
      }
    }
    __syncthreads();              // buffer p visible to all waves

    // Two k=32 sub-steps per staged slice: 32 WMMAs per barrier window.
    const int r  = lane & 15;
    const int ka = (lane >> 4) << 3;   // A: K-halves {0..7 / 8..15} (+16 for hi quad)
    const int kb = (lane >> 4) << 3;   // B: K-halves {0..7 / 8..15} (+16 for hi quad)
    #pragma unroll
    for (int c = 0; c < BK; c += 32) {
      v16h a[4], b[4];
      #pragma unroll
      for (int i = 0; i < 4; ++i) {
        const _Float16* pa = &sX[p][(wm * 64 + i * 16 + r) * BK + c + ka];
        union { v16h h; f32x4 q[2]; } u;
        u.q[0] = *(const f32x4*)(pa);
        u.q[1] = *(const f32x4*)(pa + 16);
        a[i] = u.h;
      }
      #pragma unroll
      for (int j = 0; j < 4; ++j) {
        const _Float16* pb = &sW[p][(wn * 64 + j * 16 + r) * BK + c + kb];
        union { v16h h; f32x4 q[2]; } u;
        u.q[0] = *(const f32x4*)(pb);
        u.q[1] = *(const f32x4*)(pb + 16);
        b[j] = u.h;
      }
      #pragma unroll
      for (int i = 0; i < 4; ++i)
        #pragma unroll
        for (int j = 0; j < 4; ++j)
          acc[i][j] = __builtin_amdgcn_wmma_f32_16x16x32_f16(
              false, a[i], false, b[j], (short)0, acc[i][j], false, false);
    }
  }

  // Epilogue: C/D layout — VGPR r: lanes 0-15 -> M=r, lanes 16-31 -> M=8+r; N = lane&15.
  const int cn  = lane & 15;
  const int mb8 = (lane >> 4) << 3;
  const int m0  = mblk + wm * 64;
  const int n0  = nblk + wn * 64;
  #pragma unroll
  for (int i = 0; i < 4; ++i) {
    #pragma unroll
    for (int j = 0; j < 4; ++j) {
      const int col = n0 + j * 16 + cn;
      const float bb = bias[col];
      float* yp = Y + (size_t)(m0 + i * 16 + mb8) * NDIM + col;
      #pragma unroll
      for (int q = 0; q < 8; ++q)
        yp[(size_t)q * NDIM] = acc[i][j][q] + bb;
    }
  }
}

// ================= host-side launch ========================================
extern "C" void kernel_launch(void* const* d_in, const int* in_sizes, int n_in,
                              void* d_out, int out_size, void* d_ws, size_t ws_size,
                              hipStream_t stream) {
  const float* x = (const float*)d_in[0];   // [8, 2048, 4096] fp32
  const float* W = (const float*)d_in[1];   // [4096, 4096] fp32
  const float* b = (const float*)d_in[2];   // [4096] fp32
  float* Y = (float*)d_out;                 // [8, 2048, 4096] fp32

  // Workspace: 1024 partials | alpha | pad to 8KB | Wq fp16 (32MB) | Xh fp16 (128MB)
  float*    partial = (float*)d_ws;
  float*    alpha   = partial + 1024;
  _Float16* Wq      = (_Float16*)((char*)d_ws + 8192);
  _Float16* Xh      = Wq + (size_t)NDIM * KDIM;

  k_abs_partial<<<1024, 256, 0, stream>>>(W, partial);
  k_alpha<<<1, 256, 0, stream>>>(partial, alpha);
  k_quant_w<<<(NDIM * KDIM / 4) / 256, 256, 0, stream>>>(W, alpha, Wq);
  k_cvt_x<<<(int)(((size_t)MDIM * KDIM / 4) / 256), 256, 0, stream>>>(x, Xh);

  dim3 grid(NDIM / BN, MDIM / BM);   // (16, 128)
  k_gemm<<<grid, 256, 0, stream>>>(Xh, Wq, b, Y);
}
